// TrellisNetLSTMLayer_76673756168213
// MI455X (gfx1250) — matrix operand order repaired and checked by reading, not compiled, not run on hardware
//
#include <hip/hip_runtime.h>
#include <hip/hip_bf16.h>

#define BSZ  8
#define HSZ  1024
#define SEQ  1024
#define OCH  4096      // 4*HSZ
#define KDIM 2048      // 2*HSZ

#define BM 128
#define BN 128
#define BK 64
#define NSTAGE (KDIM / BK)   // 32
#define STAGE_BYTES ((BM + BN) * BK * 2)   // 32768 per buffer

typedef __bf16 v16bf __attribute__((ext_vector_type(16)));
typedef __bf16 v8bf  __attribute__((ext_vector_type(8)));
typedef float  v8f   __attribute__((ext_vector_type(8)));

__device__ __forceinline__ unsigned short f2bf_bits(float f) {
  unsigned u = __builtin_bit_cast(unsigned, f);
  unsigned r = u + 0x7FFFu + ((u >> 16) & 1u);   // round-to-nearest-even
  return (unsigned short)(r >> 16);
}

__device__ __forceinline__ float fsig(float x) {
  return 1.0f / (1.0f + __expf(-x));
}
__device__ __forceinline__ float ftanh_(float x) {
  float e = __expf(2.0f * x);
  return 1.0f - 2.0f / (e + 1.0f);               // safe at +/- inf
}

// CDNA5 async global->LDS copy (ASYNCcnt-tracked, no VGPR data).
// GVS mode: 64-bit SGPR base + 32-bit per-lane byte offset; VDST VGPR holds
// the per-lane LDS byte address.
__device__ __forceinline__ void async_load_b128(unsigned lds, unsigned goff,
                                                const void* base) {
  asm volatile("global_load_async_to_lds_b128 %0, %1, %2"
               :: "v"(lds), "v"(goff), "s"(base) : "memory");
}
__device__ __forceinline__ void wait_async0() {
  asm volatile("s_wait_asynccnt 0x0" ::: "memory");
}

// ---------------------------------------------------------------------------
// Pack weight (4096,1024,2) fp32 -> bf16 A matrix (4096 x 2048), same layout.
// ---------------------------------------------------------------------------
__global__ void pack_weight_bf16(const float* __restrict__ w,
                                 unsigned short* __restrict__ a) {
  int idx = (blockIdx.x * blockDim.x + threadIdx.x) * 4;
  float4 f = *(const float4*)(w + idx);
  ushort4 o;
  o.x = f2bf_bits(f.x); o.y = f2bf_bits(f.y);
  o.z = f2bf_bits(f.z); o.w = f2bf_bits(f.w);
  *(ushort4*)(a + idx) = o;
}

// ---------------------------------------------------------------------------
// Pack B transposed: Bpack[b][t][2i+0] = (t==0 ? z0h[b,i] : z1h[b,i,t-1])
//                    Bpack[b][t][2i+1] = z1h[b,i,t]
// Stored N-major (t rows), K-contiguous, bf16.
// ---------------------------------------------------------------------------
__global__ void pack_b_bf16(const float* __restrict__ z1ss,
                            const float* __restrict__ z0,
                            unsigned int* __restrict__ bp) {
  int tid = blockIdx.x * blockDim.x + threadIdx.x;
  int i = tid & (HSZ - 1);
  int t = (tid >> 10) & (SEQ - 1);
  int b = tid >> 20;
  const float* z1h = z1ss + (size_t)b * (2 * HSZ) * SEQ + (size_t)i * SEQ;
  float v1 = z1h[t];
  float v0 = (t == 0) ? z0[b * (2 * HSZ) + i] : z1h[t - 1];
  unsigned int packed = (unsigned)f2bf_bits(v0) | ((unsigned)f2bf_bits(v1) << 16);
  bp[(size_t)b * SEQ * (KDIM / 2) + (size_t)t * (KDIM / 2) + i] = packed;
}

// ---------------------------------------------------------------------------
// Fused GEMM (bf16 WMMA, f32 acc) + LSTM gate epilogue.
// Workgroup tile: M=128 (= 32 channels x 4 gates), N=128, BK=64, 8 waves.
// LDS fills via global_load_async_to_lds_b128, double-buffered.
// ---------------------------------------------------------------------------
__global__ __launch_bounds__(256)
void trellis_gemm_lstm(const __bf16* __restrict__ Apack,
                       const __bf16* __restrict__ Bpack,
                       const float* __restrict__ uss,
                       const float* __restrict__ bias,
                       const float* __restrict__ z1ss,
                       const float* __restrict__ z0,
                       float* __restrict__ out) {
  __shared__ __align__(16) __bf16 smem[2 * (BM * BK + BN * BK)];  // 64 KB

  const int tid  = threadIdx.x;
  const int lane = tid & 31;
  const int wave = tid >> 5;
  const int t0 = blockIdx.x * BN;
  const int c0 = blockIdx.y * 32;
  const int b  = blockIdx.z;

  const __bf16* Bbat = Bpack + (size_t)b * SEQ * KDIM;
  const size_t ussB = (size_t)b * OCH * SEQ;
  const size_t zB   = (size_t)b * (2 * HSZ) * SEQ;

  // ---- prefetch epilogue operands into cache while the GEMM runs ----------
  {
    int tl = t0 + wave * 16;                   // 64B-aligned t line
    #pragma unroll
    for (int r = 0; r < 4; ++r) {
      int rl = r * 32 + lane;                  // 128 local rows via 4 insts
      int row = (rl >> 5) * HSZ + c0 + (rl & 31);
      __builtin_prefetch(&uss[ussB + (size_t)row * SEQ + tl], 0, 0);
    }
    int zr = HSZ + c0 + lane;                  // z1c rows for ct recurrence
    __builtin_prefetch(&z1ss[zB + (size_t)zr * SEQ + (tl ? tl - 1 : 0)], 0, 0);
  }

  v8f acc[8];
  const v8f vzero = {0.f, 0.f, 0.f, 0.f, 0.f, 0.f, 0.f, 0.f};
  #pragma unroll
  for (int i = 0; i < 8; ++i) acc[i] = vzero;

  // per-thread base offsets for async fills: row = it*32 + (tid>>3), seg = tid&7
  const unsigned ldsBase = ((unsigned)(uintptr_t)smem)   // low 32 bits == LDS addr
                         + (unsigned)((tid >> 3) * 128 + (tid & 7) * 16);
  const unsigned aBase = (unsigned)((c0 + (tid >> 3)) * (KDIM * 2) + (tid & 7) * 16);
  const unsigned bBase = (unsigned)((t0 + (tid >> 3)) * (KDIM * 2) + (tid & 7) * 16);

  auto asyncStage = [&](int buf, int k0) {
    const unsigned lb = ldsBase + (unsigned)buf * STAGE_BYTES;
    const unsigned kb = (unsigned)(k0 * 2);
    #pragma unroll
    for (int it = 0; it < 4; ++it) {
      // A rows: it*1024 + c0 + (tid>>3)  ->  +it*4MB in bytes
      async_load_b128(lb + it * 4096u,                 aBase + kb + it * 4194304u, Apack);
      // B rows: t0 + it*32 + (tid>>3)    ->  +it*128KB in bytes
      async_load_b128(lb + 16384u + it * 4096u,        bBase + kb + it * 131072u,  Bbat);
    }
  };

  auto computeStage = [&](int buf) {
    const __bf16* As = smem + buf * (BM * BK + BN * BK);
    const __bf16* Bs = As + BM * BK;
    const int nloc = wave * 16 + (lane & 15);
    const int hi   = (lane >> 4);          // half-wave select
    #pragma unroll
    for (int s = 0; s < BK; s += 32) {
      // B 32x16 bf16 frag: lanes0-15 K {0..7,16..23}, lanes16-31 K {8..15,24..31}
      const __bf16* bpp = Bs + nloc * BK + s + hi * 8;
      v8bf b0 = *(const v8bf*)(bpp);
      v8bf b1 = *(const v8bf*)(bpp + 16);
      v16bf bfrag = __builtin_shufflevector(b0, b1,
          0,1,2,3,4,5,6,7,8,9,10,11,12,13,14,15);
      #pragma unroll
      for (int mt = 0; mt < 8; ++mt) {
        // A 16x32 bf16 frag: lanes0-15 K {0..7,16..23}, lanes16-31 K {8..15,24..31}
        const __bf16* ap = As + (mt * 16 + (lane & 15)) * BK + s + hi * 8;
        v8bf a0 = *(const v8bf*)(ap);
        v8bf a1 = *(const v8bf*)(ap + 16);
        v16bf afrag = __builtin_shufflevector(a0, a1,
            0,1,2,3,4,5,6,7,8,9,10,11,12,13,14,15);
        acc[mt] = __builtin_amdgcn_wmma_f32_16x16x32_bf16(
            false, afrag, false, bfrag, (short)0, acc[mt], false, false);
      }
    }
  };

  asyncStage(0, 0);
  wait_async0();
  __syncthreads();
  for (int ks = 0; ks < NSTAGE; ++ks) {
    if (ks + 1 < NSTAGE) asyncStage((ks + 1) & 1, (ks + 1) * BK);
    computeStage(ks & 1);
    if (ks + 1 < NSTAGE) wait_async0();
    __syncthreads();
  }

  // ---------------- fused LSTM epilogue (all gates in-register) -------------
  const int t     = t0 + wave * 16 + (lane & 15);
  const int mHalf = (lane >> 4) * 8;
  #pragma unroll
  for (int cg = 0; cg < 2; ++cg) {
    #pragma unroll
    for (int v = 0; v < 8; ++v) {
      int c = c0 + cg * 16 + mHalf + v;      // C layout: VGPR v -> M = v (+8 hi)
      float pre[4];
      #pragma unroll
      for (int g = 0; g < 4; ++g) {
        int row = g * HSZ + c;
        pre[g] = acc[g * 2 + cg][v] + uss[ussB + (size_t)row * SEQ + t] + bias[row];
      }
      float it_ = fsig(pre[0]);
      float ot_ = fsig(pre[1]);
      float gt_ = ftanh_(pre[2]);
      float ft_ = fsig(pre[3]);
      float zc = (t == 0) ? z0[b * (2 * HSZ) + HSZ + c]
                          : z1ss[zB + (size_t)(HSZ + c) * SEQ + (t - 1)];
      float ct = ft_ * zc + it_ * gt_;
      float ht = ot_ * ftanh_(ct);
      out[zB + (size_t)c * SEQ + t]         = ht;
      out[zB + (size_t)(HSZ + c) * SEQ + t] = ct;
    }
  }
}

// ---------------------------------------------------------------------------
extern "C" void kernel_launch(void* const* d_in, const int* in_sizes, int n_in,
                              void* d_out, int out_size, void* d_ws, size_t ws_size,
                              hipStream_t stream) {
  const float* z1ss   = (const float*)d_in[0];
  const float* uss    = (const float*)d_in[1];
  const float* z0     = (const float*)d_in[2];
  const float* weight = (const float*)d_in[3];
  const float* bias   = (const float*)d_in[4];
  float* out = (float*)d_out;

  __bf16* Apack = (__bf16*)d_ws;                    // 16 MB
  __bf16* Bpack = Apack + (size_t)OCH * KDIM;       // +32 MB (ws total 48 MB)

  pack_weight_bf16<<<(OCH * KDIM / 4) / 256, 256, 0, stream>>>(
      weight, (unsigned short*)Apack);
  pack_b_bf16<<<(BSZ * SEQ * HSZ) / 256, 256, 0, stream>>>(
      z1ss, z0, (unsigned int*)Bpack);
  trellis_gemm_lstm<<<dim3(SEQ / BN, HSZ / 32, BSZ), 256, 0, stream>>>(
      Apack, Bpack, uss, bias, z1ss, z0, out);
}
